// simpleRNN_7859790151770
// MI455X (gfx1250) — matrix operand order repaired and checked by reading, not compiled, not run on hardware
//
#include <hip/hip_runtime.h>
#include <cmath>

typedef __attribute__((ext_vector_type(16))) _Float16 v16h;
typedef __attribute__((ext_vector_type(8)))  _Float16 v8h;
typedef __attribute__((ext_vector_type(2)))  _Float16 v2h;
typedef __attribute__((ext_vector_type(8)))  float    v8f;

#define SEQ   1024
#define BATCH 512
#define INF_  64
#define HID   256
#define OUTF  64

#define MB    32        // batch rows per workgroup
#define HPAD  264       // padded h row stride (halves): 528B -> bank-conflict free
#define XPAD  68        // padded x row stride (floats): 272B -> bank-conflict free

__device__ __forceinline__ float fast_tanh(float v) {
#if __has_builtin(__builtin_amdgcn_tanhf)
  return __builtin_amdgcn_tanhf(v);
#else
  return tanhf(v);
#endif
}

// Hidden-dim storage permutation: storage s = ng*32 + lm*2 + j  <->  logical (2ng+j)*16 + lm.
// Lets each wave store its two output columns as one packed f16 pair (b32, conflict-free).
__device__ __forceinline__ int hid_logical(int s) {
  return ((s >> 5) << 5) + ((s & 1) << 4) + ((s & 31) >> 1);
}

// One workgroup owns 32 batch rows for the entire 1024-step scan.
// 512 threads = 16 waves. Wave w: C-tile row m=w&1 (16 rows), cols n0..n0+1 (32 cols).
// V (256x256) and U (64x256) live in VGPRs in WMMA-B layout; h ping-pongs through LDS.
// x_{t+1} is async-DMA'd global->LDS during step t's compute (ASYNCcnt pipelined).
__global__ __launch_bounds__(512) void rnn_scan_kernel(
    const float* __restrict__ x, const float* __restrict__ U,
    const float* __restrict__ V, const float* __restrict__ b,
    float* __restrict__ hfinal)
{
  __shared__ _Float16 hbuf[2][MB * HPAD];   // 33.0 KB
  __shared__ float    xbuf[2][MB * XPAD];   // 17.4 KB

  const int tid  = threadIdx.x;
  const int lane = tid & 31;
  const int wave = tid >> 5;
  const int hl   = lane >> 4;        // lane half (0/1)
  const int lm   = lane & 15;
  const int m    = wave & 1;         // which 16-row tile of the 32-row batch slice
  const int ng   = wave >> 1;        // column-pair group
  const int n0   = ng * 2;           // first of two 16-col tiles
  const int r0   = blockIdx.x * MB;  // global batch row base

  // ---- V into registers, WMMA-B layout. K index goes through the storage
  //      permutation (h is stored permuted); N stays logical.
  v16h Vr[8][2];
#pragma unroll
  for (int kk = 0; kk < 8; ++kk) {
#pragma unroll
    for (int j = 0; j < 2; ++j) {
      const int N = (n0 + j) * 16 + lm;
      v16h v;
#pragma unroll
      for (int e = 0; e < 16; ++e) {
        const int sk = kk * 32 + hl * 8 + (e & 7) + ((e >> 3) << 4);       // storage (hardware) K
        v[e] = (_Float16)V[(size_t)hid_logical(sk) * HID + N];
      }
      Vr[kk][j] = v;
    }
  }
  // ---- U into registers (K = input features: NOT permuted; N logical)
  v16h Ur[2][2];
#pragma unroll
  for (int k2 = 0; k2 < 2; ++k2) {
#pragma unroll
    for (int j = 0; j < 2; ++j) {
      const int N  = (n0 + j) * 16 + lm;
      const int kb = k2 * 32 + hl * 8;
      v16h v;
#pragma unroll
      for (int e = 0; e < 16; ++e)
        v[e] = (_Float16)U[(size_t)(kb + (e & 7) + ((e >> 3) << 4)) * HID + N];
      Ur[k2][j] = v;
    }
  }
  const float bv0 = b[n0 * 16 + lm];
  const float bv1 = b[(n0 + 1) * 16 + lm];

  // h0 = 0
  for (int i = tid; i < MB * HPAD; i += 512) hbuf[0][i] = (_Float16)0.f;

  const int arow = (m * 16 + lm) * HPAD;  // A-tile row base in h buffer
  const int xrow = (m * 16 + lm) * XPAD;  // A-tile row base in x buffer
  const int xsr  = tid >> 4;              // staging: row this thread copies
  const int xsc  = (tid & 15) * 4;        // staging: col (float4)

  // ---- prologue: kick off async copy of x_0 into xbuf[0]
  {
    const unsigned long long ga =
        (unsigned long long)(x + ((size_t)(r0 + xsr)) * INF_ + xsc);
    const unsigned lva = (unsigned)(size_t)(&xbuf[0][xsr * XPAD + xsc]);
    asm volatile("global_load_async_to_lds_b128 %0, %1, off"
                 :: "v"(lva), "v"(ga) : "memory");
  }

  for (int t = 0; t < SEQ; ++t) {
    const int cur = t & 1;
    const int nxt = cur ^ 1;

    asm volatile("s_wait_asynccnt 0" ::: "memory");  // my slice of x_t landed
    __syncthreads();  // publish x_t and h_t (single barrier per step)

    // ---- issue async copy of x_{t+1} into the other buffer; it overlaps
    //      this step's compute and is waited on just before the next barrier.
    if (t + 1 < SEQ) {
      const unsigned long long ga = (unsigned long long)(
          x + ((size_t)(t + 1) * BATCH + r0 + xsr) * INF_ + xsc);
      const unsigned lva = (unsigned)(size_t)(&xbuf[nxt][xsr * XPAD + xsc]);
      asm volatile("global_load_async_to_lds_b128 %0, %1, off"
                   :: "v"(lva), "v"(ga) : "memory");
      if (t + 2 < SEQ)
        __builtin_prefetch(x + ((size_t)(t + 2) * BATCH + r0 + xsr) * INF_ + xsc, 0, 3);
    }

    // ---- acc = bias
    v8f acc0, acc1;
#pragma unroll
    for (int r = 0; r < 8; ++r) { acc0[r] = bv0; acc1[r] = bv1; }

    // ---- h @ V : 8 K-steps, A-tile software-pipelined one k ahead
    const _Float16* hc = &hbuf[cur][0];
    v16h a_cur;
    {
      v8h c0 = *reinterpret_cast<const v8h*>(hc + arow + hl * 8);
      v8h c1 = *reinterpret_cast<const v8h*>(hc + arow + hl * 8 + 16);
      a_cur = __builtin_shufflevector(c0, c1, 0, 1, 2, 3, 4, 5, 6, 7,
                                      8, 9, 10, 11, 12, 13, 14, 15);
    }
#pragma unroll
    for (int kk = 0; kk < 8; ++kk) {
      v16h a_nxt = a_cur;
      if (kk < 7) {
        const int o = arow + (kk + 1) * 32 + hl * 8;
        v8h c0 = *reinterpret_cast<const v8h*>(hc + o);
        v8h c1 = *reinterpret_cast<const v8h*>(hc + o + 16);
        a_nxt = __builtin_shufflevector(c0, c1, 0, 1, 2, 3, 4, 5, 6, 7,
                                        8, 9, 10, 11, 12, 13, 14, 15);
      }
      acc0 = __builtin_amdgcn_wmma_f32_16x16x32_f16(false, a_cur, false, Vr[kk][0],
                                                    (short)0, acc0, false, false);
      acc1 = __builtin_amdgcn_wmma_f32_16x16x32_f16(false, a_cur, false, Vr[kk][1],
                                                    (short)0, acc1, false, false);
      a_cur = a_nxt;
    }

    // ---- x_t @ U : 2 K-steps (convert f32 LDS -> f16 A on the fly)
    const float* xc = &xbuf[cur][0];
#pragma unroll
    for (int k2 = 0; k2 < 2; ++k2) {
      const int o = xrow + k2 * 32 + hl * 8;
      float4 f0 = *reinterpret_cast<const float4*>(xc + o);
      float4 f1 = *reinterpret_cast<const float4*>(xc + o + 4);
      float4 f2 = *reinterpret_cast<const float4*>(xc + o + 16);
      float4 f3 = *reinterpret_cast<const float4*>(xc + o + 20);
      v16h a;
      a[0]  = (_Float16)f0.x; a[1]  = (_Float16)f0.y;
      a[2]  = (_Float16)f0.z; a[3]  = (_Float16)f0.w;
      a[4]  = (_Float16)f1.x; a[5]  = (_Float16)f1.y;
      a[6]  = (_Float16)f1.z; a[7]  = (_Float16)f1.w;
      a[8]  = (_Float16)f2.x; a[9]  = (_Float16)f2.y;
      a[10] = (_Float16)f2.z; a[11] = (_Float16)f2.w;
      a[12] = (_Float16)f3.x; a[13] = (_Float16)f3.y;
      a[14] = (_Float16)f3.z; a[15] = (_Float16)f3.w;
      acc0 = __builtin_amdgcn_wmma_f32_16x16x32_f16(false, a, false, Ur[k2][0],
                                                    (short)0, acc0, false, false);
      acc1 = __builtin_amdgcn_wmma_f32_16x16x32_f16(false, a, false, Ur[k2][1],
                                                    (short)0, acc1, false, false);
    }

    // ---- tanh and write h_{t+1}: packed f16 pair per row -> 8 conflict-free b32 stores
    _Float16* hn = &hbuf[nxt][0];
    const int rbase = m * 16 + hl * 8;
    const int scol  = ng * 32 + lm * 2;   // storage column of the packed pair
#pragma unroll
    for (int r = 0; r < 8; ++r) {
      v2h p;
      p[0] = (_Float16)fast_tanh(acc0[r]);
      p[1] = (_Float16)fast_tanh(acc1[r]);
      *reinterpret_cast<v2h*>(&hn[(rbase + r) * HPAD + scol]) = p;
    }
    // next iteration's barrier separates these stores from their readers
  }

  __syncthreads();
  // final h lives in hbuf[SEQ & 1] == hbuf[0]; un-permute and spill as f32
  for (int i = tid; i < MB * HID; i += 512) {
    const int row = i >> 8;
    const int s   = i & 255;
    hfinal[(size_t)(r0 + row) * HID + hid_logical(s)] =
        (float)hbuf[0][row * HPAD + s];
  }
}

// out[bt][o] = h_final[bt] . W_fc[o] + b_fc[o]   (512x64 outputs, dot length 256)
__global__ __launch_bounds__(256) void fc_kernel(
    const float* __restrict__ h, const float* __restrict__ W,
    const float* __restrict__ bfc, float* __restrict__ out)
{
  const int idx = blockIdx.x * 256 + threadIdx.x;
  const int bt  = idx >> 6;
  const int o   = idx & 63;
  const float* hr = h + (size_t)bt * HID;
  const float* wr = W + (size_t)o * HID;
  float s = bfc[o];
#pragma unroll 4
  for (int k = 0; k < HID; k += 4) {
    const float4 a = *reinterpret_cast<const float4*>(hr + k);
    const float4 w = *reinterpret_cast<const float4*>(wr + k);
    s += a.x * w.x + a.y * w.y + a.z * w.z + a.w * w.w;
  }
  out[idx] = s;
}

extern "C" void kernel_launch(void* const* d_in, const int* in_sizes, int n_in,
                              void* d_out, int out_size, void* d_ws, size_t ws_size,
                              hipStream_t stream) {
  (void)in_sizes; (void)n_in; (void)out_size; (void)ws_size;
  const float* x    = (const float*)d_in[0];
  const float* U    = (const float*)d_in[1];
  const float* V    = (const float*)d_in[2];
  const float* b    = (const float*)d_in[3];
  const float* W_fc = (const float*)d_in[4];
  const float* b_fc = (const float*)d_in[5];
  float* out    = (float*)d_out;
  float* hfinal = (float*)d_ws;   // 512*256*4 = 512 KB scratch

  rnn_scan_kernel<<<BATCH / MB, 512, 0, stream>>>(x, U, V, b, hfinal);
  fc_kernel<<<(BATCH * OUTF) / 256, 256, 0, stream>>>(hfinal, W_fc, b_fc, out);
}
